// MultiHeadedAttention_7370163880352
// MI455X (gfx1250) — compile-verified, matrix operand order checked by reading, not run on hardware
//
#include <hip/hip_runtime.h>
#include <hip/hip_bf16.h>

// ---------------------------------------------------------------------------
// Multi-head attention block for MI455X (gfx1250), wave32 + WMMA f16.
//   qkv = x @ Wqkv + b ; RoPE(q,k) ; causal flash-SDPA ; out = ctx @ Wdense + b
// GEMMs are register-blocked: each wave owns a 32x64 C tile (8 WMMA / K-step).
// ---------------------------------------------------------------------------

#define S_LEN 2048
#define D_DIM 2048
#define NHEAD 16
#define HDIM  128
#define NQKV  (3 * D_DIM)

typedef _Float16 half_t;
typedef __attribute__((ext_vector_type(16))) _Float16 v16h;
typedef __attribute__((ext_vector_type(8)))  _Float16 v8h;
typedef __attribute__((ext_vector_type(8)))  float    v8f;
typedef __attribute__((ext_vector_type(4)))  float    v4f;

union V16H { v16h v; v8h h[2]; };

// Load a 16-half A/B fragment slice for one lane.
// Per ISA 16-bit layouts: halves 0..7 hold K = base..base+7,
// halves 8..15 hold K = base+16..base+23 (caller folds hsel*8 into base).
__device__ __forceinline__ v16h load_frag16(const half_t* __restrict__ base) {
  V16H u;
  u.h[0] = *(const v8h*)(base);
  u.h[1] = *(const v8h*)(base + 16);
  return u.v;
}

__device__ __forceinline__ v8f wmma32(v16h a, v16h b, v8f c) {
  // D = A(16x32 f16) * B(32x16 f16) + C(16x16 f32)
  return __builtin_amdgcn_wmma_f32_16x16x32_f16(
      /*neg_a=*/false, a, /*neg_b=*/false, b,
      /*c_mod=*/(short)0, c, /*reuse_a=*/false, /*reuse_b=*/false);
}

// Register-blocked WMMA mainloop: one wave computes C[32 x 64]
// (2 M-tiles x 4 N-tiles).  A [M,K] row-major f16, Bt [N,K] row-major f16.
// Per K-step: 2 A-frags + 4 B-frags (12x b128) feed 8 WMMAs.
__device__ __forceinline__ void gemm_block(const half_t* __restrict__ A,
                                           const half_t* __restrict__ Bt,
                                           int m0, int n0, int K,
                                           int lrow, int hsel,
                                           v8f acc[2][4]) {
  const int hoff = hsel * 8;
  const half_t* arow0 = A + (size_t)(m0 + lrow) * K;
  const half_t* arow1 = arow0 + (size_t)16 * K;
  const half_t* brow0 = Bt + (size_t)(n0 + lrow) * K;
  const half_t* brow1 = brow0 + (size_t)16 * K;
  const half_t* brow2 = brow0 + (size_t)32 * K;
  const half_t* brow3 = brow0 + (size_t)48 * K;

#pragma unroll 2
  for (int k0 = 0; k0 < K; k0 += 32) {
    __builtin_prefetch(arow0 + k0 + 512, 0, 1);   // global_prefetch_b8
    __builtin_prefetch(brow0 + k0 + 512, 0, 1);
    v16h a0 = load_frag16(arow0 + k0 + hoff);
    v16h a1 = load_frag16(arow1 + k0 + hoff);
    v16h b0 = load_frag16(brow0 + k0 + hoff);
    v16h b1 = load_frag16(brow1 + k0 + hoff);
    v16h b2 = load_frag16(brow2 + k0 + hoff);
    v16h b3 = load_frag16(brow3 + k0 + hoff);
    acc[0][0] = wmma32(a0, b0, acc[0][0]);
    acc[1][0] = wmma32(a1, b0, acc[1][0]);
    acc[0][1] = wmma32(a0, b1, acc[0][1]);
    acc[1][1] = wmma32(a1, b1, acc[1][1]);
    acc[0][2] = wmma32(a0, b2, acc[0][2]);
    acc[1][2] = wmma32(a1, b2, acc[1][2]);
    acc[0][3] = wmma32(a0, b3, acc[0][3]);
    acc[1][3] = wmma32(a1, b3, acc[1][3]);
  }
}

// ---------------------------------------------------------------------------
// Elementwise conversion kernels
// ---------------------------------------------------------------------------
__global__ void cvt_f32_f16_kernel(const float* __restrict__ src,
                                   half_t* __restrict__ dst, int n) {
  int i = blockIdx.x * blockDim.x + threadIdx.x;
  if (i < n) dst[i] = (half_t)src[i];
}

// LDS-tiled transpose: src [K,N] f32 row-major -> dst [N,K] f16 row-major.
// 32x32 tile per 256-thread block; coalesced loads AND stores.
__global__ __launch_bounds__(256)
void cvt_transpose_kernel(const float* __restrict__ src,
                          half_t* __restrict__ dst, int K, int N) {
  __shared__ half_t tile[32][33];          // +1 pad: kill bank conflicts
  const int k0 = blockIdx.y * 32;
  const int n0 = blockIdx.x * 32;
  const int tx = threadIdx.x & 31;
  const int ty = threadIdx.x >> 5;         // 0..7
#pragma unroll
  for (int i = 0; i < 32; i += 8) {
    tile[ty + i][tx] = (half_t)src[(size_t)(k0 + ty + i) * N + n0 + tx];
  }
  __syncthreads();
#pragma unroll
  for (int i = 0; i < 32; i += 8) {
    dst[(size_t)(n0 + ty + i) * K + k0 + tx] = tile[tx][ty + i];
  }
}

// ---------------------------------------------------------------------------
// Fused QKV projection: qkv = xh @ WqT^T + bqkv
//   Q,K -> fp32 [S, D] (pre-RoPE)   V -> f16 transposed [H*HD, S]
// Block = 256 threads (8 waves, 2x4 wave grid) => 64 x 256 C per block.
// ---------------------------------------------------------------------------
__global__ __launch_bounds__(256)
void gemm_qkv_kernel(const half_t* __restrict__ xh,
                     const half_t* __restrict__ WqT,
                     const float* __restrict__ bqkv,
                     float* __restrict__ Qf, float* __restrict__ Kf,
                     half_t* __restrict__ Vt) {
  const int lane = threadIdx.x & 31;
  const int wave = threadIdx.x >> 5;       // 0..7
  const int wm = wave & 1;                 // 2 waves over M
  const int wn = wave >> 1;                // 4 waves over N
  const int lrow = lane & 15;
  const int hsel = lane >> 4;
  const int m0 = blockIdx.y * 64 + wm * 32;
  const int n0 = blockIdx.x * 256 + wn * 64;

  v8f acc[2][4];
#pragma unroll
  for (int mi = 0; mi < 2; ++mi)
#pragma unroll
    for (int ni = 0; ni < 4; ++ni) { v8f z = {}; acc[mi][ni] = z; }

  gemm_block(xh, WqT, m0, n0, D_DIM, lrow, hsel, acc);

#pragma unroll
  for (int ni = 0; ni < 4; ++ni) {
    const int n = n0 + ni * 16 + lrow;     // global output column [0, 6144)
    const float bias = bqkv[n];
#pragma unroll
    for (int mi = 0; mi < 2; ++mi) {
#pragma unroll
      for (int r = 0; r < 8; ++r) {
        const int row = m0 + mi * 16 + r + 8 * hsel;
        const float val = acc[mi][ni][r] + bias;
        if (n < D_DIM) {
          Qf[(size_t)row * D_DIM + n] = val;
        } else if (n < 2 * D_DIM) {
          Kf[(size_t)row * D_DIM + (n - D_DIM)] = val;
        } else {
          const int c = n - 2 * D_DIM;     // = h*128 + d
          Vt[(size_t)c * S_LEN + row] = (half_t)val;   // [H*HD, S]
        }
      }
    }
  }
}

// ---------------------------------------------------------------------------
// RoPE on Q and K; also re-lay out to f16 [H, S, HD].
// ---------------------------------------------------------------------------
__global__ void rope_kernel(const float* __restrict__ Qf,
                            const float* __restrict__ Kf,
                            const float* __restrict__ fc,   // [S, HD/2, 2]
                            half_t* __restrict__ Qh, half_t* __restrict__ Kh) {
  int idx = blockIdx.x * blockDim.x + threadIdx.x;
  const int NP = S_LEN * NHEAD * (HDIM / 2);
  if (idx >= NP) return;
  const int i = idx % (HDIM / 2);
  const int h = (idx / (HDIM / 2)) % NHEAD;
  const int s = idx / ((HDIM / 2) * NHEAD);
  const float c  = fc[((size_t)s * (HDIM / 2) + i) * 2 + 0];
  const float sn = fc[((size_t)s * (HDIM / 2) + i) * 2 + 1];
  const size_t src = (size_t)s * D_DIM + h * HDIM + 2 * i;
  const float q0 = Qf[src], q1 = Qf[src + 1];
  const float k0 = Kf[src], k1 = Kf[src + 1];
  const size_t dst = ((size_t)h * S_LEN + s) * HDIM + 2 * i;
  Qh[dst]     = (half_t)(q0 * c - q1 * sn);
  Qh[dst + 1] = (half_t)(q1 * c + q0 * sn);
  Kh[dst]     = (half_t)(k0 * c - k1 * sn);
  Kh[dst + 1] = (half_t)(k1 * c + k0 * sn);
}

// ---------------------------------------------------------------------------
// Causal flash attention: one wave per (head, 16-query tile).
//   scores = (Q Kᵀ)/sqrt(HD); online softmax; ctx += P V
// Q/K: f16 [H,S,HD]; V: f16 [H*HD, S]; ctx out: f16 [S, D].
// P repack (C-layout f32 -> A-layout f16) goes through LDS.
// ---------------------------------------------------------------------------
__global__ __launch_bounds__(32)
void attn_kernel(const half_t* __restrict__ Qh, const half_t* __restrict__ Kh,
                 const half_t* __restrict__ Vt, half_t* __restrict__ ctxh) {
  __shared__ float sP[16 * 32];

  const int lane = threadIdx.x & 31;
  const int lrow = lane & 15;
  const int hsel = lane >> 4;
  const int hoff = hsel * 8;
  const int head = blockIdx.x;
  const int q0 = blockIdx.y * 16;

  const half_t* Qbase = Qh + (size_t)head * S_LEN * HDIM;
  const half_t* Kbase = Kh + (size_t)head * S_LEN * HDIM;
  const half_t* Vbase = Vt + (size_t)head * HDIM * S_LEN;

  // Q fragments for the 4 K-chunks of HD=128 (kept in registers).
  v16h aq[4];
  const half_t* qrow = Qbase + (size_t)(q0 + lrow) * HDIM;
#pragma unroll
  for (int j = 0; j < 4; ++j) aq[j] = load_frag16(qrow + j * 32 + hoff);

  v8f ctx[8];
#pragma unroll
  for (int t = 0; t < 8; ++t) { v8f z = {}; ctx[t] = z; }
  float m[8], l[8];
#pragma unroll
  for (int r = 0; r < 8; ++r) { m[r] = -1e30f; l[r] = 0.0f; }
  const float scale = 0.08838834764831845f;   // 1/sqrt(128)

  const int kend = q0 + 16;                   // causal: keys <= q0+15
  for (int k0 = 0; k0 < kend; k0 += 32) {
    // ---- scores: two 16x16 f32 tiles over keys [k0, k0+32) ----
    v8f st[2];
#pragma unroll
    for (int t = 0; t < 2; ++t) {
      v8f s = {};
      const half_t* krow = Kbase + (size_t)(k0 + t * 16 + lrow) * HDIM;
#pragma unroll
      for (int j = 0; j < 4; ++j) {
        v16h bk = load_frag16(krow + j * 32 + hoff);
        s = wmma32(aq[j], bk, s);
      }
      st[t] = s;
    }
    // ---- scale + causal mask + row max (reduce across 16 lanes) ----
    float rmax[8];
#pragma unroll
    for (int r = 0; r < 8; ++r) {
      const int qi = r + 8 * hsel;            // row within tile
#pragma unroll
      for (int t = 0; t < 2; ++t) {
        const int key = k0 + t * 16 + lrow;
        float v = st[t][r] * scale;
        if (key > q0 + qi) v = -1e30f;
        st[t][r] = v;
      }
      float v = fmaxf(st[0][r], st[1][r]);
      for (int off = 1; off < 16; off <<= 1) v = fmaxf(v, __shfl_xor(v, off, 16));
      rmax[r] = v;
    }
    // ---- online softmax update ----
#pragma unroll
    for (int r = 0; r < 8; ++r) {
      const float mnew = fmaxf(m[r], rmax[r]);
      const float cf = __expf(m[r] - mnew);
      m[r] = mnew;
      l[r] *= cf;
#pragma unroll
      for (int t = 0; t < 2; ++t) st[t][r] = __expf(st[t][r] - mnew);
      float rs = st[0][r] + st[1][r];
      for (int off = 1; off < 16; off <<= 1) rs += __shfl_xor(rs, off, 16);
      l[r] += rs;
#pragma unroll
      for (int nt = 0; nt < 8; ++nt) ctx[nt][r] *= cf;
    }
    // ---- repack P (16x32 f32, C-layout) -> f16 A-fragment via LDS ----
    __syncthreads();
#pragma unroll
    for (int r = 0; r < 8; ++r) {
      const int qi = r + 8 * hsel;
      sP[qi * 32 + lrow]      = st[0][r];
      sP[qi * 32 + 16 + lrow] = st[1][r];
    }
    __syncthreads();
    v16h pa;
    {
      const float* prow = sP + lrow * 32 + hoff;
      v4f p0 = *(const v4f*)(prow);
      v4f p1 = *(const v4f*)(prow + 4);
      v4f p2 = *(const v4f*)(prow + 16);
      v4f p3 = *(const v4f*)(prow + 20);
      V16H u;
#pragma unroll
      for (int i = 0; i < 4; ++i) {
        u.h[0][i]     = (half_t)p0[i];
        u.h[0][i + 4] = (half_t)p1[i];
        u.h[1][i]     = (half_t)p2[i];
        u.h[1][i + 4] = (half_t)p3[i];
      }
      pa = u.v;
    }
    // ---- ctx += P(16x32) @ V(32x128) : 8 WMMA tiles over the dims ----
#pragma unroll
    for (int nt = 0; nt < 8; ++nt) {
      const half_t* vrow = Vbase + (size_t)(nt * 16 + lrow) * S_LEN + k0 + hoff;
      v16h bv = load_frag16(vrow);
      ctx[nt] = wmma32(pa, bv, ctx[nt]);
    }
  }
  // ---- normalize and store ctx as f16 [S, D] ----
#pragma unroll
  for (int r = 0; r < 8; ++r) {
    const int row = q0 + r + 8 * hsel;
    const float inv = 1.0f / l[r];
#pragma unroll
    for (int nt = 0; nt < 8; ++nt) {
      ctxh[(size_t)row * D_DIM + head * HDIM + nt * 16 + lrow] =
          (half_t)(ctx[nt][r] * inv);
    }
  }
}

// ---------------------------------------------------------------------------
// Output projection: out = ctxh @ WdT^T + bdense  (fp32 out)
// ---------------------------------------------------------------------------
__global__ __launch_bounds__(256)
void gemm_out_kernel(const half_t* __restrict__ ctxh,
                     const half_t* __restrict__ WdT,
                     const float* __restrict__ bdense,
                     float* __restrict__ out) {
  const int lane = threadIdx.x & 31;
  const int wave = threadIdx.x >> 5;
  const int wm = wave & 1;
  const int wn = wave >> 1;
  const int lrow = lane & 15;
  const int hsel = lane >> 4;
  const int m0 = blockIdx.y * 64 + wm * 32;
  const int n0 = blockIdx.x * 256 + wn * 64;

  v8f acc[2][4];
#pragma unroll
  for (int mi = 0; mi < 2; ++mi)
#pragma unroll
    for (int ni = 0; ni < 4; ++ni) { v8f z = {}; acc[mi][ni] = z; }

  gemm_block(ctxh, WdT, m0, n0, D_DIM, lrow, hsel, acc);

#pragma unroll
  for (int ni = 0; ni < 4; ++ni) {
    const int n = n0 + ni * 16 + lrow;
    const float bias = bdense[n];
#pragma unroll
    for (int mi = 0; mi < 2; ++mi) {
#pragma unroll
      for (int r = 0; r < 8; ++r) {
        const int row = m0 + mi * 16 + r + 8 * hsel;
        out[(size_t)row * D_DIM + n] = acc[mi][ni][r] + bias;
      }
    }
  }
}

// ---------------------------------------------------------------------------
// Host launcher
// ---------------------------------------------------------------------------
extern "C" void kernel_launch(void* const* d_in, const int* in_sizes, int n_in,
                              void* d_out, int out_size, void* d_ws, size_t ws_size,
                              hipStream_t stream) {
  const float* x      = (const float*)d_in[0];
  const float* fc     = (const float*)d_in[1];
  // d_in[2] = input_pos (identity layout assumed, matching the reference)
  const float* Wqkv   = (const float*)d_in[3];
  const float* bqkv   = (const float*)d_in[4];
  const float* Wdense = (const float*)d_in[5];
  const float* bdense = (const float*)d_in[6];
  float* out = (float*)d_out;

  // Workspace layout (bytes)
  char* ws = (char*)d_ws;
  const size_t XH_B   = (size_t)S_LEN * D_DIM * 2;          //  8 MB  x f16
  const size_t WQT_B  = (size_t)NQKV * D_DIM * 2;           // 25 MB  WqkvT f16
  const size_t WDT_B  = (size_t)D_DIM * D_DIM * 2;          //  8 MB  WdenseT f16
  const size_t QF_B   = (size_t)S_LEN * D_DIM * 4;          // 16 MB  Q f32
  const size_t KF_B   = (size_t)S_LEN * D_DIM * 4;          // 16 MB  K f32
  const size_t VT_B   = (size_t)NHEAD * HDIM * S_LEN * 2;   //  8 MB  Vt f16
  const size_t QH_B   = (size_t)NHEAD * S_LEN * HDIM * 2;   //  8 MB
  const size_t KH_B   = QH_B;                               //  8 MB

  half_t* xh   = (half_t*)(ws);
  half_t* WqT  = (half_t*)(ws + XH_B);
  half_t* WdT  = (half_t*)(ws + XH_B + WQT_B);
  float*  Qf   = (float*) (ws + XH_B + WQT_B + WDT_B);
  float*  Kf   = (float*) (ws + XH_B + WQT_B + WDT_B + QF_B);
  half_t* Vt   = (half_t*)(ws + XH_B + WQT_B + WDT_B + QF_B + KF_B);
  half_t* Qh   = (half_t*)(ws + XH_B + WQT_B + WDT_B + QF_B + KF_B + VT_B);
  half_t* Kh   = (half_t*)(ws + XH_B + WQT_B + WDT_B + QF_B + KF_B + VT_B + QH_B);
  half_t* ctxh = (half_t*)(ws + XH_B + WQT_B + WDT_B + QF_B + KF_B + VT_B + QH_B + KH_B);

  // 1) precision conversion / weight transpose (LDS-tiled, coalesced)
  {
    int n = S_LEN * D_DIM;
    cvt_f32_f16_kernel<<<(n + 255) / 256, 256, 0, stream>>>(x, xh, n);
  }
  cvt_transpose_kernel<<<dim3(NQKV / 32, D_DIM / 32), 256, 0, stream>>>(Wqkv, WqT, D_DIM, NQKV);
  cvt_transpose_kernel<<<dim3(D_DIM / 32, D_DIM / 32), 256, 0, stream>>>(Wdense, WdT, D_DIM, D_DIM);

  // 2) fused QKV projection (WMMA, 32x64 C per wave)
  gemm_qkv_kernel<<<dim3(NQKV / 256, S_LEN / 64), 256, 0, stream>>>(xh, WqT, bqkv, Qf, Kf, Vt);

  // 3) RoPE
  {
    int n = S_LEN * NHEAD * (HDIM / 2);
    rope_kernel<<<(n + 255) / 256, 256, 0, stream>>>(Qf, Kf, fc, Qh, Kh);
  }

  // 4) causal flash SDPA (WMMA)
  attn_kernel<<<dim3(NHEAD, S_LEN / 16), 32, 0, stream>>>(Qh, Kh, Vt, ctxh);

  // 5) output projection (WMMA, 32x64 C per wave)
  gemm_out_kernel<<<dim3(D_DIM / 256, S_LEN / 64), 256, 0, stream>>>(ctxh, WdT, bdense, out);
}
